// RNN_Model_78082505441614
// MI455X (gfx1250) — matrix operand order repaired and checked by reading, not compiled, not run on hardware
//
#include <hip/hip_runtime.h>
#include <math.h>

// ---------------------------------------------------------------------------
// Problem dims (fixed by the reference)
// ---------------------------------------------------------------------------
#define QQ 512
#define NN 4096
#define HH 1024
#define LL 8
#define G4H (4 * HH)   // 4096

typedef unsigned short u16;
typedef unsigned int   u32;
typedef __bf16  v16bf __attribute__((ext_vector_type(16)));
typedef float   v8f   __attribute__((ext_vector_type(8)));
typedef unsigned uv4  __attribute__((ext_vector_type(4)));
typedef int      iv4  __attribute__((ext_vector_type(4)));
typedef int      iv8  __attribute__((ext_vector_type(8)));

__device__ __forceinline__ float bf2f(u16 u) {
    return __uint_as_float(((u32)u) << 16);
}
__device__ __forceinline__ u16 f2bf(float f) {
    u32 x = __float_as_uint(f);
    u32 r = x + 0x7FFFu + ((x >> 16) & 1u);   // round-to-nearest-even
    return (u16)(r >> 16);
}

// ---------------------------------------------------------------------------
// fp32 -> bf16 conversion (one-time; makes the ~150 MB weight set L2-resident)
// ---------------------------------------------------------------------------
__global__ void cvt_f32_bf16(const float* __restrict__ in, u16* __restrict__ out, int n) {
    int i = blockIdx.x * blockDim.x + threadIdx.x;
    int stride = gridDim.x * blockDim.x;
    for (; i < n; i += stride) out[i] = f2bf(in[i]);
}

// ---------------------------------------------------------------------------
// WMMA GEMM:  C[M,Ncols] = A[M,K] * Bw[Ncols,K]^T + bias1(+bias2), opt sigmoid
// Each wave computes a 16(M) x 64(N) strip: one A fragment feeds 4 WMMAs
// (4:10 wmma:b128-load ratio). 8 waves/block cover 16 x 512.
//
// Fragment layouts used here (wave32, same K order for A and B):
//  A 16x32 bf16 : lane L holds row M=L&15; lanes 0-15 carry K[0..15],
//                 lanes 16-31 carry K[16..31]  -> two b128 loads.
//  B 32x16 bf16 : lane L holds col N=L&15; lanes 0-15 carry K[0..15],
//                 lanes 16-31 carry K[16..31]           -> two b128 loads.
//  C 16x16 f32  : VGPR r: M=r (lanes 0-15) / M=r+8 (lanes 16-31), N=lane&15.
// ---------------------------------------------------------------------------
template <int ACT>   // 0 = none, 1 = sigmoid
__global__ void wmma_gemm_bt(const u16* __restrict__ A,
                             const u16* __restrict__ Bw,
                             const float* __restrict__ bias1,
                             const float* __restrict__ bias2,
                             float* __restrict__ Cf,     // f32 out (or null)
                             u16*   __restrict__ Cbf,    // bf16 out (or null)
                             int M, int Ncols, int K) {
    const int wave  = threadIdx.x >> 5;
    const int lane  = threadIdx.x & 31;
    const int half  = lane >> 4;       // 0 or 1
    const int mrow  = lane & 15;
    const int tileM = blockIdx.y * 16;
    const int tileN = blockIdx.x * 512 + wave * 64;

    v8f acc0 = {}, acc1 = {}, acc2 = {}, acc3 = {};

    const u16* arow = A  + (size_t)(tileM + mrow) * K;
    const u16* b0   = Bw + (size_t)(tileN +  0 + mrow) * K;
    const u16* b1   = Bw + (size_t)(tileN + 16 + mrow) * K;
    const u16* b2   = Bw + (size_t)(tileN + 32 + mrow) * K;
    const u16* b3   = Bw + (size_t)(tileN + 48 + mrow) * K;

    union Frag { uv4 q[2]; v16bf v; };
    for (int kb = 0; kb < K; kb += 32) {
        Frag fa, f0, f1, f2, f3;
        const u16* ap = arow + kb + half * 16;
        fa.q[0] = *(const uv4*)(ap);
        fa.q[1] = *(const uv4*)(ap + 8);
        const int ko = kb + half * 16;
        f0.q[0] = *(const uv4*)(b0 + ko); f0.q[1] = *(const uv4*)(b0 + ko + 8);
        f1.q[0] = *(const uv4*)(b1 + ko); f1.q[1] = *(const uv4*)(b1 + ko + 8);
        f2.q[0] = *(const uv4*)(b2 + ko); f2.q[1] = *(const uv4*)(b2 + ko + 8);
        f3.q[0] = *(const uv4*)(b3 + ko); f3.q[1] = *(const uv4*)(b3 + ko + 8);

        acc0 = __builtin_amdgcn_wmma_f32_16x16x32_bf16(false, fa.v, false, f0.v, (short)0, acc0, false, false);
        acc1 = __builtin_amdgcn_wmma_f32_16x16x32_bf16(false, fa.v, false, f1.v, (short)0, acc1, false, false);
        acc2 = __builtin_amdgcn_wmma_f32_16x16x32_bf16(false, fa.v, false, f2.v, (short)0, acc2, false, false);
        acc3 = __builtin_amdgcn_wmma_f32_16x16x32_bf16(false, fa.v, false, f3.v, (short)0, acc3, false, false);
    }

    const int mbase = tileM + half * 8;
    v8f accs[4] = {acc0, acc1, acc2, acc3};
#pragma unroll
    for (int s = 0; s < 4; ++s) {
        const int nIdx = tileN + s * 16 + mrow;
        float b = bias1 ? bias1[nIdx] : 0.0f;
        if (bias2) b += bias2[nIdx];
#pragma unroll
        for (int r = 0; r < 8; ++r) {
            float v = accs[s][r] + b;
            if (ACT == 1) v = 1.0f / (1.0f + __expf(-v));
            size_t idx = (size_t)(mbase + r) * Ncols + nIdx;
            if (Cf)  Cf[idx]  = v;
            if (Cbf) Cbf[idx] = f2bf(v);
        }
    }
}

// ---------------------------------------------------------------------------
// Persistent LSTM scan for one layer.
//  - 64 WGs x 256 threads; WG b owns 16 hidden units -> 64 gate rows.
//  - The WG's W_hh slab (64 rows x 1024 bf16 = 128 KB) is loaded into LDS
//    ONCE via the Tensor Data Mover (2D tile: 4 gate blocks of 16 contiguous
//    rows, row-block stride H*H) and reused for all 512 timesteps — zero
//    global weight traffic on the serial critical path.
//  - Per step, h_{t-1} (4 KB) is pulled into LDS with async-to-LDS loads.
//  - 4 threads per gate row each dot 256 of K from LDS; LDS reduction; 16
//    threads apply the cell nonlinearity with c held in registers.
//  - h broadcast through a global f32 buffer + grid-wide atomic spin barrier.
// ---------------------------------------------------------------------------
#define SCAN_UNITS 16                        // hidden units per WG
#define SCAN_ROWS  (4 * SCAN_UNITS)          // 64 gate rows per WG
#define SCAN_WGS   (HH / SCAN_UNITS)         // 64 workgroups
#define SCAN_SMEM  (SCAN_ROWS * HH * 2 + HH * 4 + 256 * 4 + SCAN_ROWS * 4)

__global__ void lstm_scan(const u16*  __restrict__ Whh,    // [4H, H] bf16
                          const float* __restrict__ Gpre,  // [Q, 4H] (biases folded in)
                          u16*   __restrict__ Xnext,       // [Q, H] bf16 h outputs
                          float* __restrict__ hG,          // [H] f32 h broadcast
                          u32*   __restrict__ ctr,         // grid barrier (zeroed)
                          float* __restrict__ out_h,       // [H] final h
                          float* __restrict__ out_c) {     // [H] final c
    extern __shared__ char smem[];
    u16*   w_lds = (u16*)smem;                               // 128 KB weights
    float* h_sh  = (float*)(smem + (size_t)SCAN_ROWS * HH * 2);
    float* red   = h_sh + HH;
    float* g_sh  = red + 256;

    const int tid  = threadIdx.x;
    const int lr   = tid >> 2;          // 0..63 local gate row = gate*16 + uloc
    const int q    = tid & 3;           // K quarter
    const int gate = lr >> 4;
    const int uloc = lr & 15;
    const int base = blockIdx.x * SCAN_UNITS;
    const int grow = gate * HH + base + uloc;   // global gate row

    // ---- one-time TDM load of this WG's W_hh slab into LDS ----
    if (tid < 32) {
        unsigned long long ga = (unsigned long long)(const void*)(Whh + (size_t)base * HH);
        u32 lds_base = (u32)(unsigned long long)(void*)w_lds;
        const u32 dim0  = (u32)SCAN_UNITS * HH;   // 16384 elems per gate block
        const u32 strd0 = (u32)HH * HH;           // stride between gate blocks

        uv4 g0;
        g0[0] = 1u;                                           // count=1 (valid D#)
        g0[1] = lds_base;                                     // lds_addr
        g0[2] = (u32)ga;                                      // global_addr[31:0]
        g0[3] = (u32)((ga >> 32) & 0x01FFFFFFu) | (2u << 30); // global_addr[56:32] | type=2

        iv8 g1;
        g1[0] = (int)(1u << 16);                              // data_size=2B; mask=0
        g1[1] = (int)((dim0 & 0xFFFFu) << 16);                // tensor_dim0[15:0]
        g1[2] = (int)((dim0 >> 16) | (4u << 16));             // tensor_dim0[31:16] | tensor_dim1=4
        g1[3] = (int)((dim0 & 0xFFFFu) << 16);                // tile_dim0 = dim0
        g1[4] = (int)(4u);                                    // tile_dim1=4, tile_dim2=0
        g1[5] = (int)(strd0);                                 // tensor_dim0_stride[31:0]
        g1[6] = 0;                                            // stride0[47:32]=0
        g1[7] = 0;

        iv4 gz4 = {0, 0, 0, 0};
        iv8 gz8 = {0, 0, 0, 0, 0, 0, 0, 0};
        __builtin_amdgcn_tensor_load_to_lds(g0, g1, gz4, gz4, gz8, 0);
        __builtin_amdgcn_s_wait_tensorcnt(0);
    }
    __syncthreads();

    float c_reg = 0.0f;

    for (int t = 0; t < QQ; ++t) {
        // ---- stage h_{t-1} into LDS (async-to-LDS path; zeros at t==0) ----
        if (t == 0) {
            for (int k = tid; k < HH; k += 256) h_sh[k] = 0.0f;
        } else {
            for (int k = tid; k < HH; k += 256) {
                u32 lo = (u32)(unsigned long long)(void*)(h_sh + k);
                asm volatile("global_load_async_to_lds_b32 %0, %1, off"
                             :: "v"(lo), "v"(hG + k) : "memory");
            }
            asm volatile("s_wait_asynccnt 0x0" ::: "memory");
        }
        __syncthreads();

        if (q == 0 && t + 1 < QQ)
            __builtin_prefetch(Gpre + (size_t)(t + 1) * G4H + grow, 0, 0);

        // ---- partial dot from LDS-resident weights ----
        const u16*   wl = w_lds + (size_t)lr * HH + q * 256;
        const float* hp = h_sh + q * 256;
        float part = 0.0f;
#pragma unroll 4
        for (int k = 0; k < 256; k += 8) {
            uv4 w = *(const uv4*)(wl + k);
#pragma unroll
            for (int p = 0; p < 4; ++p) {
                u32 d = w[p];
                part += bf2f((u16)(d & 0xFFFFu)) * hp[k + 2 * p];
                part += bf2f((u16)(d >> 16))     * hp[k + 2 * p + 1];
            }
        }
        red[tid] = part;
        __syncthreads();

        if (q == 0) {
            g_sh[lr] = red[tid] + red[tid + 1] + red[tid + 2] + red[tid + 3]
                     + Gpre[(size_t)t * G4H + grow];
        }
        __syncthreads();

        // ---- cell nonlinearity; c carried in registers of threads 0..15 ----
        if (tid < SCAN_UNITS) {
            float gi = g_sh[tid];
            float gf = g_sh[SCAN_UNITS + tid];
            float gg = g_sh[2 * SCAN_UNITS + tid];
            float go = g_sh[3 * SCAN_UNITS + tid];
            gi = 1.0f / (1.0f + __expf(-gi));
            gf = 1.0f / (1.0f + __expf(-gf));
            go = 1.0f / (1.0f + __expf(-go));
            c_reg = gf * c_reg + gi * tanhf(gg);
            float h = go * tanhf(c_reg);
            hG[base + tid] = h;
            Xnext[(size_t)t * HH + base + tid] = f2bf(h);
            if (t == QQ - 1) { out_h[base + tid] = h; out_c[base + tid] = c_reg; }
        }
        __syncthreads();

        // ---- grid-wide barrier: publish h, wait for all WGs ----
        if (tid == 0) {
            __threadfence();
            __hip_atomic_fetch_add(ctr, 1u, __ATOMIC_ACQ_REL, __HIP_MEMORY_SCOPE_AGENT);
            const u32 target = (u32)(SCAN_WGS * (t + 1));
            while (__hip_atomic_load(ctr, __ATOMIC_ACQUIRE, __HIP_MEMORY_SCOPE_AGENT) < target) {
                __builtin_amdgcn_s_sleep(1);
            }
        }
        __syncthreads();
    }
}

// ---------------------------------------------------------------------------
// Host-side orchestration
// ---------------------------------------------------------------------------
extern "C" void kernel_launch(void* const* d_in, const int* in_sizes, int n_in,
                              void* d_out, int out_size, void* d_ws, size_t ws_size,
                              hipStream_t stream) {
    const float* inputs = (const float*)d_in[0];   // [Q,N]
    const float* W_inp  = (const float*)d_in[1];   // [H,N]
    const float* b_inp  = (const float*)d_in[2];   // [H]
    const float* W_ih   = (const float*)d_in[3];   // [L,4H,H]
    const float* W_hh   = (const float*)d_in[4];   // [L,4H,H]
    const float* b_ih   = (const float*)d_in[5];   // [L,4H]
    const float* b_hh   = (const float*)d_in[6];   // [L,4H]
    const float* W_lin  = (const float*)d_in[7];   // [N,H]
    const float* b_lin  = (const float*)d_in[8];   // [N]
    float* out = (float*)d_out;                    // [Q*N] ++ [L*H] ++ [L*H]

    // workspace carve-up (bytes)
    char* ws = (char*)d_ws;
    size_t off = 0;
    auto take = [&](size_t bytes) { char* p = ws + off; off += (bytes + 255) & ~(size_t)255; return p; };
    u16* inputs_bf = (u16*)take((size_t)QQ * NN * 2);        //   4 MB
    u16* Winp_bf   = (u16*)take((size_t)HH * NN * 2);        //   8 MB
    u16* Wih_bf    = (u16*)take((size_t)LL * G4H * HH * 2);  //  64 MB
    u16* Whh_bf    = (u16*)take((size_t)LL * G4H * HH * 2);  //  64 MB
    u16* Wlin_bf   = (u16*)take((size_t)NN * HH * 2);        //   8 MB
    u16* X0        = (u16*)take((size_t)QQ * HH * 2);        //   1 MB
    u16* X1        = (u16*)take((size_t)QQ * HH * 2);        //   1 MB
    float* Gpre    = (float*)take((size_t)QQ * G4H * 4);     //   8 MB
    float* hG      = (float*)take((size_t)HH * 4);
    u32*  ctr      = (u32*)take(256);

    // allow 133 KB of dynamic LDS for the persistent scan (WGP has 320 KB)
    (void)hipFuncSetAttribute((const void*)lstm_scan,
                              hipFuncAttributeMaxDynamicSharedMemorySize, SCAN_SMEM);

    // 1) one-time bf16 conversion of activations + all weights
    const int CB = 2048, CT = 256;
    cvt_f32_bf16<<<CB, CT, 0, stream>>>(inputs, inputs_bf, QQ * NN);
    cvt_f32_bf16<<<CB, CT, 0, stream>>>(W_inp,  Winp_bf,   HH * NN);
    cvt_f32_bf16<<<CB, CT, 0, stream>>>(W_ih,   Wih_bf,    LL * G4H * HH);
    cvt_f32_bf16<<<CB, CT, 0, stream>>>(W_hh,   Whh_bf,    LL * G4H * HH);
    cvt_f32_bf16<<<CB, CT, 0, stream>>>(W_lin,  Wlin_bf,   NN * HH);

    // 2) input projection: X0[Q,H] = inputs @ W_inp^T + b_inp   (WMMA)
    wmma_gemm_bt<0><<<dim3(HH / 512, QQ / 16), 256, 0, stream>>>(
        inputs_bf, Winp_bf, b_inp, nullptr, nullptr, X0, QQ, HH, NN);

    // 3) per layer: hoisted W_ih GEMM (WMMA) + sequential W_hh scan
    u16* Xc = X0;
    u16* Xn = X1;
    for (int l = 0; l < LL; ++l) {
        wmma_gemm_bt<0><<<dim3(G4H / 512, QQ / 16), 256, 0, stream>>>(
            Xc, Wih_bf + (size_t)l * G4H * HH,
            b_ih + (size_t)l * G4H, b_hh + (size_t)l * G4H,
            Gpre, nullptr, QQ, G4H, HH);

        (void)hipMemsetAsync(ctr, 0, sizeof(u32), stream);
        lstm_scan<<<SCAN_WGS, 256, SCAN_SMEM, stream>>>(
            Whh_bf + (size_t)l * G4H * HH, Gpre, Xn, hG, ctr,
            out + (size_t)QQ * NN + (size_t)l * HH,
            out + (size_t)QQ * NN + (size_t)LL * HH + (size_t)l * HH);

        u16* tmp = Xc; Xc = Xn; Xn = tmp;
    }

    // 4) output projection + sigmoid: out[Q,N] = sigmoid(Xc @ W_lin^T + b_lin)
    wmma_gemm_bt<1><<<dim3(NN / 512, QQ / 16), 256, 0, stream>>>(
        Xc, Wlin_bf, b_lin, nullptr, out, nullptr, QQ, NN, HH);
}
